// TransformerBlock_11458972745860
// MI455X (gfx1250) — matrix operand order repaired and checked by reading, not compiled, not run on hardware
//
#include <hip/hip_runtime.h>
#include <hip/hip_bf16.h>
#include <math.h>

// ---------------------------------------------------------------------------
// Types
// ---------------------------------------------------------------------------
typedef __bf16 bf16_t;
typedef __attribute__((ext_vector_type(16))) __bf16 v16bf;
typedef __attribute__((ext_vector_type(8)))  __bf16 v8bf;
typedef __attribute__((ext_vector_type(8)))  float  v8f;
typedef __attribute__((ext_vector_type(4)))  unsigned v4u;
typedef __attribute__((ext_vector_type(8)))  int      v8i;
typedef __attribute__((ext_vector_type(4)))  int      v4i;

#define D_MODEL   2048
#define NUM_HEADS 16
#define D_FF      8192
#define SEQ       2048
#define BATCH     2
#define BT        (BATCH * SEQ)          // 4096 tokens
#define DK        (D_MODEL / NUM_HEADS)  // 128
#define F3        (3 * D_MODEL)          // 6144
#define EPSV      1e-6f

#if __has_builtin(__builtin_amdgcn_tensor_load_to_lds)
#define HAVE_TDM 1
#else
#define HAVE_TDM 0
#endif

// ---------------------------------------------------------------------------
// bf16 <-> f32 helpers (bit-exact RNE)
// ---------------------------------------------------------------------------
static __device__ __forceinline__ bf16_t f2bf(float f) {
    unsigned u = __builtin_bit_cast(unsigned, f);
    unsigned r = (u + 0x7FFFu + ((u >> 16) & 1u)) >> 16;
    unsigned short s = (unsigned short)r;
    return __builtin_bit_cast(bf16_t, s);
}
static __device__ __forceinline__ float bf2f(bf16_t b) {
    unsigned short s = __builtin_bit_cast(unsigned short, b);
    unsigned u = ((unsigned)s) << 16;
    return __builtin_bit_cast(float, u);
}

// ---------------------------------------------------------------------------
// CDNA5 async global->LDS copy (ASYNCcnt-tracked) — fallback staging path.
// LDS destination = low 32 bits of the flat shared-memory address
// (ISA 10.2: LDS aperture addresses truncate to the LDS byte offset).
// ---------------------------------------------------------------------------
static __device__ __forceinline__ void async_copy_b128(unsigned lds_off,
                                                       const void* gsrc) {
    asm volatile("global_load_async_to_lds_b128 %0, %1, off"
                 :: "v"(lds_off), "v"(gsrc) : "memory");
}
static __device__ __forceinline__ void wait_asynccnt0() {
    asm volatile("s_wait_asynccnt 0" ::: "memory");
}
static __device__ __forceinline__ unsigned lds_offset(const void* shared_ptr) {
    return (unsigned)(size_t)shared_ptr;
}

// ---------------------------------------------------------------------------
// WMMA fragment loaders, matching CDNA5 ISA 7.12.2 VGPR layouts (wave32).
// A (16x32, 16-bit): lanes 0-15 = rows 0-15 with K {0..7,16..23},
//                    lanes 16-31 = rows 0-15 with K {8..15,24..31}.
// B (32x16, 16-bit): lanes 0-15 = cols 0-15 with K {0..7,16..23},
//                    lanes 16-31 = cols 0-15 with K {8..15,24..31}.
// ---------------------------------------------------------------------------
static __device__ __forceinline__ v16bf load_afrag(const bf16_t* base, int ld) {
    const int lane = threadIdx.x & 31;
    const int row  = lane & 15;
    const int koff = (lane < 16) ? 0 : 8;
    v8bf a0 = *(const v8bf*)(base + (size_t)row * ld + koff);
    v8bf a1 = *(const v8bf*)(base + (size_t)row * ld + koff + 16);
    v16bf a;
#pragma unroll
    for (int i = 0; i < 8; ++i) { a[i] = a0[i]; a[i + 8] = a1[i]; }
    return a;
}
static __device__ __forceinline__ v16bf load_bfrag(const bf16_t* base, int ld) {
    const int lane = threadIdx.x & 31;
    const int col  = lane & 15;
    const int koff = (lane < 16) ? 0 : 8;
    v8bf b0 = *(const v8bf*)(base + (size_t)col * ld + koff);
    v8bf b1 = *(const v8bf*)(base + (size_t)col * ld + koff + 16);
    v16bf b;
#pragma unroll
    for (int i = 0; i < 8; ++i) { b[i] = b0[i]; b[i + 8] = b1[i]; }
    return b;
}
static __device__ __forceinline__ v8f wmma_bf16(v16bf a, v16bf b, v8f c) {
    return __builtin_amdgcn_wmma_f32_16x16x32_bf16(false, a, false, b,
                                                   (short)0, c, false, false);
}

// ---------------------------------------------------------------------------
// fp32 -> bf16 conversion (weights)
// ---------------------------------------------------------------------------
__launch_bounds__(256)
__global__ void cvt_bf16_kernel(const float* __restrict__ src,
                                bf16_t* __restrict__ dst, size_t n) {
    size_t i = (size_t)blockIdx.x * blockDim.x + threadIdx.x;
    size_t stride = (size_t)gridDim.x * blockDim.x;
    for (; i < n; i += stride) dst[i] = f2bf(src[i]);
}

// ---------------------------------------------------------------------------
// RMSNorm (block per row) + bf16 cast
// ---------------------------------------------------------------------------
__launch_bounds__(256)
__global__ void rmsnorm_bf16_kernel(const float* __restrict__ x,
                                    const float* __restrict__ g,
                                    bf16_t* __restrict__ out) {
    __shared__ float red[8];
    const int row = blockIdx.x;
    const float* xr = x + (size_t)row * D_MODEL;
    float ss = 0.f;
    for (int i = threadIdx.x; i < D_MODEL; i += 256) {
        float v = xr[i];
        ss += v * v;
    }
#pragma unroll
    for (int m = 1; m < 32; m <<= 1) ss += __shfl_xor(ss, m, 32);
    if ((threadIdx.x & 31) == 0) red[threadIdx.x >> 5] = ss;
    __syncthreads();
    if (threadIdx.x == 0) {
        float s = 0.f;
#pragma unroll
        for (int i = 0; i < 8; ++i) s += red[i];
        red[0] = s;
    }
    __syncthreads();
    const float inv = rsqrtf(red[0] / (float)D_MODEL + EPSV);
    for (int i = threadIdx.x; i < D_MODEL; i += 256)
        out[(size_t)row * D_MODEL + i] = f2bf(xr[i] * inv * g[i]);
}

// ---------------------------------------------------------------------------
// Tiled bf16 NT GEMM:  C[M,N] (+epilogue) = A[M,K] * B[N,K]^T
//   MODE 1: store bf16           -> outB
//   MODE 2: store fp32 res + acc -> outF (fused residual add)
// Block: 256 threads (8 waves). Tile 128x128x32, double-buffered LDS staged
// with the CDNA5 Tensor Data Mover (TENSORcnt) when available, else with
// async global->LDS copies (ASYNCcnt). LDS rows padded (ld=40) to spread
// fragment reads across banks; the TDM reproduces the padding in hardware
// via D#.pad_interval/pad_amount (16 dwords data + 4 dwords pad per row).
// ---------------------------------------------------------------------------
#define BM 128
#define BN 128
#define BK 32
#define LDA 40   // padded LDS row stride (bf16 elements), 80B per row

template <int MODE>
__launch_bounds__(256)
__global__ void gemm_nt_kernel(const bf16_t* __restrict__ A,
                               const bf16_t* __restrict__ B,
                               int M, int N, int K,
                               const float* __restrict__ res,
                               float* __restrict__ outF,
                               bf16_t* __restrict__ outB) {
    // +8 elements guard: TDM emits one pad group after the final row too.
    __shared__ bf16_t sA[2][BM * LDA + 8];
    __shared__ bf16_t sB[2][BN * LDA + 8];

    const int tid  = threadIdx.x;
    const int n0   = blockIdx.x * BN;
    const int m0   = blockIdx.y * BM;
    const int wave = tid >> 5;
    const int wm   = (wave >> 2) * 64;   // 0 / 64
    const int wn   = (wave & 3) * 32;    // 0 / 32 / 64 / 96
    const int srow = tid >> 1;           // 0..127 staging row
    const int scol = (tid & 1) * 16;     // 0 / 16  staging col

    const bf16_t* gA = A + (size_t)(m0 + srow) * K + scol;
    const bf16_t* gB = B + (size_t)(n0 + srow) * K + scol;

    v8f acc[4][2];
#pragma unroll
    for (int mi = 0; mi < 4; ++mi)
#pragma unroll
        for (int ni = 0; ni < 2; ++ni) acc[mi][ni] = (v8f){};

#if HAVE_TDM
    // Tensor DMA descriptor group 1 (ISA 8.4), shared by A and B tiles:
    //   data_size=1 (2B), pad_enable, pad_interval=3 (16 dwords),
    //   pad_amount=3 (4 dwords), tensor_dim0=K, tensor_dim1=128,
    //   tile_dim0=32, tile_dim1=128, tensor_dim0_stride=K.
    v8i tg1 = {};
    tg1[0] = (int)((1u << 16) | (1u << 20) | (3u << 22) | (3u << 25));
    tg1[1] = (int)(((unsigned)K & 0xFFFFu) << 16);
    tg1[2] = (int)((((unsigned)K >> 16) & 0xFFFFu) | (128u << 16));
    tg1[3] = (int)(32u << 16);
    tg1[4] = 128;
    tg1[5] = K;
    const v4i z4 = {};
    const v8i z8 = {};
    auto tdm_issue = [&](unsigned lds, const bf16_t* g) {
        unsigned long long ga = (unsigned long long)(size_t)g;
        v4u g0;
        g0[0] = 1u;                                   // count=1 (valid D#)
        g0[1] = lds;                                  // lds_addr
        g0[2] = (unsigned)ga;                         // global_addr[31:0]
        g0[3] = (unsigned)((ga >> 32) & 0x01FFFFFFu)  // global_addr[56:32]
                | (2u << 30);                         // type=2 ("image")
#if __clang_major__ >= 23
        __builtin_amdgcn_tensor_load_to_lds(g0, tg1, z4, z4, z8, 0);
#else
        __builtin_amdgcn_tensor_load_to_lds(g0, tg1, z4, z4, 0);
#endif
    };
    auto stage = [&](int buf, int k0) {
        if (tid < 32) {  // wave 0 issues both tile DMAs
            tdm_issue(lds_offset(&sA[buf][0]), A + (size_t)m0 * K + k0);
            tdm_issue(lds_offset(&sB[buf][0]), B + (size_t)n0 * K + k0);
        }
    };
    auto stage_wait = [&]() {
        if (tid < 32) __builtin_amdgcn_s_wait_tensorcnt((short)0);
    };
#else
    // Fallback: 4 async b128 copies per thread per stage (ASYNCcnt).
    auto stage = [&](int buf, int k0) {
        const unsigned la = lds_offset(&sA[buf][srow * LDA + scol]);
        const unsigned lb = lds_offset(&sB[buf][srow * LDA + scol]);
        async_copy_b128(la,      gA + k0);
        async_copy_b128(la + 16, gA + k0 + 8);
        async_copy_b128(lb,      gB + k0);
        async_copy_b128(lb + 16, gB + k0 + 8);
    };
    auto stage_wait = [&]() { wait_asynccnt0(); };
#endif

    const int nk = K / BK;
    stage(0, 0);
    stage_wait();
    __syncthreads();

    for (int kb = 0; kb < nk; ++kb) {
        const int cur = kb & 1;
        if (kb + 2 < nk) {
            __builtin_prefetch(gA + (kb + 2) * BK, 0, 1);  // global_prefetch_b8
            __builtin_prefetch(gB + (kb + 2) * BK, 0, 1);
        }
        if (kb + 1 < nk) stage(cur ^ 1, (kb + 1) * BK);

        v16bf bfr0 = load_bfrag(&sB[cur][(wn + 0)  * LDA], LDA);
        v16bf bfr1 = load_bfrag(&sB[cur][(wn + 16) * LDA], LDA);
#pragma unroll
        for (int mi = 0; mi < 4; ++mi) {
            v16bf af = load_afrag(&sA[cur][(wm + mi * 16) * LDA], LDA);
            acc[mi][0] = wmma_bf16(af, bfr0, acc[mi][0]);
            acc[mi][1] = wmma_bf16(af, bfr1, acc[mi][1]);
        }
        stage_wait();   // next-stage copies have landed in LDS
        __syncthreads();
    }

    // Writeback: C/D layout — VGPR r, lane l => (M = r + 8*(l>>4), N = l&15)
    const int lane = tid & 31;
    const int cN   = lane & 15;
    const int rM   = (lane >> 4) * 8;
#pragma unroll
    for (int mi = 0; mi < 4; ++mi)
#pragma unroll
        for (int ni = 0; ni < 2; ++ni) {
            const int gm = m0 + wm + mi * 16 + rM;
            const int gn = n0 + wn + ni * 16 + cN;
#pragma unroll
            for (int r = 0; r < 8; ++r) {
                const size_t idx = (size_t)(gm + r) * N + gn;
                const float v = acc[mi][ni][r];
                if (MODE == 1) outB[idx] = f2bf(v);
                else           outF[idx] = res[idx] + v;
            }
        }
}

// ---------------------------------------------------------------------------
// Extract V^T: vt[(b*H+h), c, t] = qkv[t_global, 2C + h*128 + c]   (bf16)
// ---------------------------------------------------------------------------
__launch_bounds__(256)
__global__ void split_v_kernel(const bf16_t* __restrict__ qkv,
                               bf16_t* __restrict__ vt) {
    size_t i = (size_t)blockIdx.x * blockDim.x + threadIdx.x;
    const size_t n = (size_t)BT * D_MODEL;
    const size_t stride = (size_t)gridDim.x * blockDim.x;
    for (; i < n; i += stride) {
        const int t  = (int)(i / D_MODEL);
        const int ch = (int)(i % D_MODEL);
        const int h = ch >> 7, c = ch & 127;
        const int b = t / SEQ, tt = t % SEQ;
        vt[((size_t)(b * NUM_HEADS + h) * DK + c) * SEQ + tt] =
            qkv[(size_t)t * F3 + 2 * D_MODEL + ch];
    }
}

// ---------------------------------------------------------------------------
// Causal flash attention. One wave (32 threads) per (b, h, 16-query tile).
// Q,K read from qkv bf16 buffer (contiguous along dk), V from transposed vt.
// ---------------------------------------------------------------------------
__launch_bounds__(32)
__global__ void flash_attn_kernel(const bf16_t* __restrict__ qkv,
                                  const bf16_t* __restrict__ vt,
                                  bf16_t* __restrict__ out) {
    __shared__ bf16_t sP[16 * 32];

    const int lane = threadIdx.x & 31;
    const int nqt  = SEQ / 16;
    const int qt   = blockIdx.x % nqt;
    const int bh   = blockIdx.x / nqt;
    const int b    = bh / NUM_HEADS;
    const int h    = bh % NUM_HEADS;
    const int q0   = qt * 16;
    const float scale = 0.08838834764831845f;  // 1/sqrt(128)

    // Q fragments: 4 chunks of 16x32 along dk
    const bf16_t* qbase = qkv + (size_t)(b * SEQ + q0) * F3 + h * DK;
    v16bf qf[4];
#pragma unroll
    for (int c = 0; c < 4; ++c) qf[c] = load_afrag(qbase + c * 32, F3);

    float mrow[8], lrow[8], alpha[8];
#pragma unroll
    for (int r = 0; r < 8; ++r) { mrow[r] = -3.0e38f; lrow[r] = 0.f; }
    v8f acc[8];
#pragma unroll
    for (int n = 0; n < 8; ++n) acc[n] = (v8f){};

    const int rOff = (lane >> 4) * 8;   // row group of this lane
    const int nkb  = (q0 + 47) >> 5;    // causal: key blocks of 32

    for (int kb = 0; kb < nkb; ++kb) {
        const int kbase = kb * 32;
        // S = Q * K^T over this 32-key block (two 16x16 N-subtiles)
        const bf16_t* kb0 = qkv + (size_t)(b * SEQ + kbase) * F3 + D_MODEL + h * DK;
        const bf16_t* kb1 = kb0 + (size_t)16 * F3;
        v8f s0 = (v8f){}, s1 = (v8f){};
#pragma unroll
        for (int c = 0; c < 4; ++c) {
            s0 = wmma_bf16(qf[c], load_bfrag(kb0 + c * 32, F3), s0);
            s1 = wmma_bf16(qf[c], load_bfrag(kb1 + c * 32, F3), s1);
        }
        // Online softmax (row stats live per (lane, r); reduce within 16-lane halves)
#pragma unroll
        for (int r = 0; r < 8; ++r) {
            const int m   = q0 + r + rOff;
            const int k0i = kbase + (lane & 15);
            float x0 = s0[r] * scale; if (k0i      > m) x0 = -3.0e38f;
            float x1 = s1[r] * scale; if (k0i + 16 > m) x1 = -3.0e38f;
            float t = fmaxf(x0, x1);
            t = fmaxf(t, __shfl_xor(t, 1, 32));
            t = fmaxf(t, __shfl_xor(t, 2, 32));
            t = fmaxf(t, __shfl_xor(t, 4, 32));
            t = fmaxf(t, __shfl_xor(t, 8, 32));
            const float mnew = fmaxf(mrow[r], t);
            const float a    = __expf(mrow[r] - mnew);
            const float p0   = __expf(x0 - mnew);
            const float p1   = __expf(x1 - mnew);
            float ps = p0 + p1;
            ps += __shfl_xor(ps, 1, 32);
            ps += __shfl_xor(ps, 2, 32);
            ps += __shfl_xor(ps, 4, 32);
            ps += __shfl_xor(ps, 8, 32);
            lrow[r] = lrow[r] * a + ps;
            mrow[r] = mnew;
            alpha[r] = a;
            s0[r] = p0; s1[r] = p1;
        }
#pragma unroll
        for (int n = 0; n < 8; ++n)
#pragma unroll
            for (int r = 0; r < 8; ++r) acc[n][r] *= alpha[r];

        // Stage P (16x32 bf16) via LDS to re-layout into an A-fragment
        __syncthreads();
#pragma unroll
        for (int r = 0; r < 8; ++r) {
            const int m = r + rOff;
            sP[m * 32 + (lane & 15)]      = f2bf(s0[r]);
            sP[m * 32 + 16 + (lane & 15)] = f2bf(s1[r]);
        }
        __syncthreads();
        v16bf pf = load_afrag(sP, 32);

        // O += P * V  (V^T rows contiguous along keys)
#pragma unroll
        for (int n = 0; n < 8; ++n) {
            const bf16_t* vb = vt + ((size_t)bh * DK + n * 16) * SEQ + kbase;
            acc[n] = wmma_bf16(pf, load_bfrag(vb, SEQ), acc[n]);
        }
    }

    // Epilogue: divide by row sums, write bf16 [B,T,H*dk]
#pragma unroll
    for (int n = 0; n < 8; ++n)
#pragma unroll
        for (int r = 0; r < 8; ++r) {
            const int m  = q0 + r + rOff;
            const int ch = h * DK + n * 16 + (lane & 15);
            out[(size_t)(b * SEQ + m) * D_MODEL + ch] = f2bf(acc[n][r] / lrow[r]);
        }
}

// ---------------------------------------------------------------------------
// gu = silu(gate) * up (bf16 in/out, in-place into gate buffer)
// ---------------------------------------------------------------------------
__launch_bounds__(256)
__global__ void silu_mul_kernel(bf16_t* __restrict__ gate,
                                const bf16_t* __restrict__ up, size_t n) {
    size_t i = (size_t)blockIdx.x * blockDim.x + threadIdx.x;
    const size_t stride = (size_t)gridDim.x * blockDim.x;
    for (; i < n; i += stride) {
        const float g = bf2f(gate[i]);
        const float u = bf2f(up[i]);
        const float s = g / (1.f + __expf(-g));
        gate[i] = f2bf(s * u);
    }
}

// ---------------------------------------------------------------------------
// Host orchestration
// ---------------------------------------------------------------------------
static size_t align256(size_t x) { return (x + 255) & ~(size_t)255; }

extern "C" void kernel_launch(void* const* d_in, const int* in_sizes, int n_in,
                              void* d_out, int out_size, void* d_ws, size_t ws_size,
                              hipStream_t stream) {
    const float* x      = (const float*)d_in[0];
    const float* w_qkv  = (const float*)d_in[1];
    const float* w_o    = (const float*)d_in[2];
    const float* g1     = (const float*)d_in[3];
    const float* g2     = (const float*)d_in[4];
    const float* w_gate = (const float*)d_in[5];
    const float* w_up   = (const float*)d_in[6];
    const float* w_down = (const float*)d_in[7];
    float* out = (float*)d_out;

    // Workspace carve-up (~420 MB total)
    char* p = (char*)d_ws;
    size_t off = 0;
    auto take = [&](size_t bytes) { char* r = p + off; off = align256(off + bytes); return r; };
    bf16_t* xn     = (bf16_t*)take((size_t)BT * D_MODEL * 2);
    bf16_t* wqkv_b = (bf16_t*)take((size_t)F3 * D_MODEL * 2);
    bf16_t* qkv    = (bf16_t*)take((size_t)BT * F3 * 2);
    bf16_t* vtb    = (bf16_t*)take((size_t)BATCH * NUM_HEADS * DK * SEQ * 2);
    bf16_t* attn   = (bf16_t*)take((size_t)BT * D_MODEL * 2);
    bf16_t* wo_b   = (bf16_t*)take((size_t)D_MODEL * D_MODEL * 2);
    float*  x1     = (float* )take((size_t)BT * D_MODEL * 4);
    bf16_t* xn2    = (bf16_t*)take((size_t)BT * D_MODEL * 2);
    bf16_t* wg_b   = (bf16_t*)take((size_t)D_FF * D_MODEL * 2);
    bf16_t* wu_b   = (bf16_t*)take((size_t)D_FF * D_MODEL * 2);
    bf16_t* wd_b   = (bf16_t*)take((size_t)D_MODEL * D_FF * 2);
    bf16_t* gate   = (bf16_t*)take((size_t)BT * D_FF * 2);
    bf16_t* up     = (bf16_t*)take((size_t)BT * D_FF * 2);
    (void)ws_size; (void)n_in; (void)in_sizes; (void)out_size;

    // 1) Weights -> bf16
    cvt_bf16_kernel<<<4096, 256, 0, stream>>>(w_qkv,  wqkv_b, (size_t)F3 * D_MODEL);
    cvt_bf16_kernel<<<4096, 256, 0, stream>>>(w_o,    wo_b,   (size_t)D_MODEL * D_MODEL);
    cvt_bf16_kernel<<<4096, 256, 0, stream>>>(w_gate, wg_b,   (size_t)D_FF * D_MODEL);
    cvt_bf16_kernel<<<4096, 256, 0, stream>>>(w_up,   wu_b,   (size_t)D_FF * D_MODEL);
    cvt_bf16_kernel<<<4096, 256, 0, stream>>>(w_down, wd_b,   (size_t)D_MODEL * D_FF);

    // 2) xn = rmsnorm(x, g1) -> bf16
    rmsnorm_bf16_kernel<<<BT, 256, 0, stream>>>(x, g1, xn);

    // 3) qkv = xn @ w_qkv^T  -> bf16 [BT, 6144]
    gemm_nt_kernel<1><<<dim3(F3 / BN, BT / BM), 256, 0, stream>>>(
        xn, wqkv_b, BT, F3, D_MODEL, nullptr, nullptr, qkv);

    // 4) V^T extraction
    split_v_kernel<<<4096, 256, 0, stream>>>(qkv, vtb);

    // 5) causal flash attention -> attn bf16 [BT, C]
    flash_attn_kernel<<<BATCH * NUM_HEADS * (SEQ / 16), 32, 0, stream>>>(qkv, vtb, attn);

    // 6) x1 = x + attn @ w_o^T  (fused residual)
    gemm_nt_kernel<2><<<dim3(D_MODEL / BN, BT / BM), 256, 0, stream>>>(
        attn, wo_b, BT, D_MODEL, D_MODEL, x, x1, nullptr);

    // 7) xn2 = rmsnorm(x1, g2) -> bf16
    rmsnorm_bf16_kernel<<<BT, 256, 0, stream>>>(x1, g2, xn2);

    // 8,9) gate / up GEMMs -> bf16 [BT, FF]
    gemm_nt_kernel<1><<<dim3(D_FF / BN, BT / BM), 256, 0, stream>>>(
        xn2, wg_b, BT, D_FF, D_MODEL, nullptr, nullptr, gate);
    gemm_nt_kernel<1><<<dim3(D_FF / BN, BT / BM), 256, 0, stream>>>(
        xn2, wu_b, BT, D_FF, D_MODEL, nullptr, nullptr, up);

    // 10) gu = silu(gate) * up  (in place)
    silu_mul_kernel<<<8192, 256, 0, stream>>>(gate, up, (size_t)BT * D_FF);

    // 11) out = x1 + gu @ w_down^T  (fused residual, fp32 output)
    gemm_nt_kernel<2><<<dim3(D_MODEL / BN, BT / BM), 256, 0, stream>>>(
        gate, wd_b, BT, D_MODEL, D_FF, x1, out, nullptr);
}
